// DenseSAKELayer_71116068487535
// MI455X (gfx1250) — compile-verified, matrix operand order checked by reading, not run on hardware
//
#include <hip/hip_runtime.h>
#include <hip/hip_bf16.h>

#define BB 4
#define NN 512
#define FF 64
#define EPS 1e-14f

typedef __attribute__((ext_vector_type(16))) __bf16 v16bf;
typedef __attribute__((ext_vector_type(8)))  float  v8f;

// Fast TRANS32-based math (v_rcp_f32 / v_sqrt_f32 / v_exp_f32 co-execute
// with the XDL WMMA pipe on CDNA5; avoids 10-op IEEE divide sequences).
__device__ __forceinline__ float fast_rcp(float v)  { return __builtin_amdgcn_rcpf(v); }
__device__ __forceinline__ float fast_sqrt(float v) { return __builtin_amdgcn_sqrtf(v); }
__device__ __forceinline__ float fast_silu(float z) {
  return z * fast_rcp(1.f + __expf(-z));
}
__device__ __forceinline__ float fast_tanh(float z) {
  const float zc = fminf(fmaxf(z, -15.f), 15.f);
  const float e = __expf(2.f * zc);
  return (e - 1.f) * fast_rcp(e + 1.f);
}

// ---------------------------------------------------------------------------
// Stage 1: per-node projections (tiny GEMMs, 34M MACs total -> plain FMA).
//   pj/pi : h @ W_ew[:F] / h @ W_ew[F:]          [B*N,64]
//   ej/ei : h @ W_es[1:F+1] / h @ W_es[F+1:]     [B*N,64]
//   qj/qi : h @ W_sa[:F] / h @ W_sa[F:]          [B*N]
// ---------------------------------------------------------------------------
__global__ __launch_bounds__(64) void sake_node_proj(
    const float* __restrict__ h, const float* __restrict__ W_ew,
    const float* __restrict__ W_sa, const float* __restrict__ W_es,
    float* __restrict__ pj, float* __restrict__ pi,
    float* __restrict__ ej, float* __restrict__ ei,
    float* __restrict__ qj, float* __restrict__ qi) {
  __shared__ float sh[64];
  __shared__ float sred[64];
  const int n = blockIdx.x;          // 0 .. B*N-1
  const int t = threadIdx.x;         // 64 threads
  sh[t] = h[n * 64 + t];
  __syncthreads();
  float a0 = 0.f, a1 = 0.f, a2 = 0.f, a3 = 0.f;
#pragma unroll 8
  for (int f = 0; f < 64; ++f) {
    const float hv = sh[f];
    a0 += hv * W_ew[f * 64 + t];
    a1 += hv * W_ew[(64 + f) * 64 + t];
    a2 += hv * W_es[(1 + f) * 64 + t];
    a3 += hv * W_es[(65 + f) * 64 + t];
  }
  pj[n * 64 + t] = a0;
  pi[n * 64 + t] = a1;
  ej[n * 64 + t] = a2;
  ei[n * 64 + t] = a3;
  // qj / qi: dot(h, W_sa[:64]) and dot(h, W_sa[64:])
  sred[t] = sh[t] * W_sa[t];
  __syncthreads();
  for (int s = 32; s > 0; s >>= 1) {
    if (t < s) sred[t] += sred[t + s];
    __syncthreads();
  }
  if (t == 0) qj[n] = sred[0];
  __syncthreads();
  sred[t] = sh[t] * W_sa[64 + t];
  __syncthreads();
  for (int s = 32; s > 0; s >>= 1) {
    if (t < s) sred[t] += sred[t + s];
    __syncthreads();
  }
  if (t == 0) qi[n] = sred[0];
}

// ---------------------------------------------------------------------------
// Stage 2: fused pair kernel. One workgroup (128 thr = 4 waves) per (b,i).
// Each wave processes 8 chunks of 16 j's. Per chunk:
//   - softmax weight w_j = leakyrelu(qj+qi) * exp(norm - max)/denom
//   - h_e[16x64] = silu(filtered*W_es0 + ej + ei + b_es) -> LDS
//   - agg[f]  += w_j * h_e            (h_e_agg)
//   - att[k,c]+= tanh(pj+pi+b_ew) * v (einsum)
//   - T = h_e @ W_c1 via 8x v_wmma_f32_16x16x32_bf16 (W_c1 frags in VGPRs)
//   - coord = silu(T+b_c1) @ W_c2 + b_c2 ; x accum += coord * xmx
// ---------------------------------------------------------------------------
__global__ __launch_bounds__(128) void sake_main(
    const float* __restrict__ h, const float* __restrict__ x,
    const float* __restrict__ b_ew, const float* __restrict__ W_es,
    const float* __restrict__ b_es, const float* __restrict__ W_c1,
    const float* __restrict__ b_c1, const float* __restrict__ W_c2,
    const float* __restrict__ b_c2, const float* __restrict__ W_n,
    const float* __restrict__ b_n,
    const float* __restrict__ pj, const float* __restrict__ pi,
    const float* __restrict__ ej, const float* __restrict__ ei,
    const float* __restrict__ qj, const float* __restrict__ qi,
    float* __restrict__ out) {
  __shared__ float s_x[NN * 3];        // x[b,:,:]
  __shared__ float s_Wc1[64 * 64];     // W_c1 staged for fragment build
  __shared__ float s_he[4 * 16 * 66];  // per-wave h_e tile, pitch 66 (bank-safe)
  __shared__ float s_w[64];            // per-wave softmax*sem weights
  __shared__ float s_coord[64];        // per-wave coord per j
  __shared__ float s_pi[64], s_ei[64], s_bew[64], s_bes[64];
  __shared__ float s_bc1[64], s_wc2[64], s_hi[64], s_wes0[64];
  __shared__ float s_aggW[4 * 64];     // per-wave h_e_agg partials
  __shared__ float s_attW[4 * 192];    // per-wave att_sum partials [k][c]
  __shared__ float s_axW[4 * 48];      // per-wave x-update partials
  __shared__ float s_agg[64], s_attn[64];
  __shared__ float s_red[128];

  const int blk = blockIdx.x;
  const int b = blk / NN;
  const int i = blk - b * NN;
  const int t = threadIdx.x;
  const int lane = t & 31;
  const int wave = t >> 5;
  const int lo = lane & 15;
  const int hi = lane >> 4;
  const int node = b * NN + i;

  // ---- stage shared data -------------------------------------------------
  for (int idx = t; idx < NN * 3; idx += 128) s_x[idx] = x[b * NN * 3 + idx];
  for (int idx = t; idx < 64 * 64; idx += 128) s_Wc1[idx] = W_c1[idx];
  if (t < 64) {
    s_pi[t] = pi[(size_t)node * 64 + t];
    s_ei[t] = ei[(size_t)node * 64 + t];
    s_bew[t] = b_ew[t];
    s_bes[t] = b_es[t];
    s_bc1[t] = b_c1[t];
    s_wc2[t] = W_c2[t];
    s_hi[t] = h[(size_t)node * 64 + t];
    s_wes0[t] = W_es[t];  // row 0 of W_es
  }
  __syncthreads();

  const float xi0 = s_x[i * 3 + 0], xi1 = s_x[i * 3 + 1], xi2 = s_x[i * 3 + 2];
  const float qii = qi[node];
  const float bc2v = b_c2[0];

  // ---- softmax over j of norm: max then sum(exp) -------------------------
  float lmax = -1e30f;
  for (int j = t; j < NN; j += 128) {
    const float dx = s_x[j * 3 + 0] - xi0;
    const float dy = s_x[j * 3 + 1] - xi1;
    const float dz = s_x[j * 3 + 2] - xi2;
    lmax = fmaxf(lmax, fast_sqrt(dx * dx + dy * dy + dz * dz + EPS));
  }
  s_red[t] = lmax;
  __syncthreads();
  for (int s = 64; s > 0; s >>= 1) {
    if (t < s) s_red[t] = fmaxf(s_red[t], s_red[t + s]);
    __syncthreads();
  }
  const float smax = s_red[0];
  __syncthreads();
  float lsum = 0.f;
  for (int j = t; j < NN; j += 128) {
    const float dx = s_x[j * 3 + 0] - xi0;
    const float dy = s_x[j * 3 + 1] - xi1;
    const float dz = s_x[j * 3 + 2] - xi2;
    lsum += __expf(fast_sqrt(dx * dx + dy * dy + dz * dz + EPS) - smax);
  }
  s_red[t] = lsum;
  __syncthreads();
  for (int s = 64; s > 0; s >>= 1) {
    if (t < s) s_red[t] += s_red[t + s];
    __syncthreads();
  }
  const float invden = fast_rcp(s_red[0]);
  __syncthreads();

  // ---- build W_c1 B-fragments (kept in VGPRs for all 8 chunks) -----------
  // B layout mirrors A: lane -> n = 16*tile + lo, khigh = hi; element e:
  // k = 32*kstep + ((e/2<4)?0:16) + 2*((e/2)&3) + 8*hi + (e&1)
  v16bf Bf[4][2];
#pragma unroll
  for (int tt = 0; tt < 4; ++tt)
#pragma unroll
    for (int ss = 0; ss < 2; ++ss)
#pragma unroll
      for (int e = 0; e < 16; ++e) {
        const int r = e >> 1;
        const int kk = ((r < 4) ? 0 : 16) + 2 * (r & 3) + 8 * hi + (e & 1) + 32 * ss;
        Bf[tt][ss][e] = (__bf16)s_Wc1[kk * 64 + 16 * tt + lo];
      }

  // ---- main loop over j-chunks -------------------------------------------
  float agg0 = 0.f, agg1 = 0.f;
  float att00 = 0.f, att01 = 0.f, att02 = 0.f;
  float att10 = 0.f, att11 = 0.f, att12 = 0.f;
  float ax0 = 0.f, ax1 = 0.f, ax2 = 0.f;
  const int f0 = 2 * lane;
  const float* ejB = ej + (size_t)b * NN * 64;
  const float* pjB = pj + (size_t)b * NN * 64;
  const float* qjB = qj + b * NN;
  float* sheW = s_he + wave * (16 * 66);

  for (int it = 0; it < 8; ++it) {
    const int jbase = (wave + it * 4) * 16;

    // prefetch next chunk's ej/pj rows into cache (global_prefetch_b8)
    if (it < 7) {
      const int jn = jbase + 64 + lo;  // next chunk for this wave
      __builtin_prefetch(ejB + (size_t)jn * 64 + 32 * hi, 0, 1);
      __builtin_prefetch(pjB + (size_t)jn * 64 + 32 * hi, 0, 1);
    }

    // phase A: softmax*semantic weights for this chunk (lanes 0-15)
    if (lane < 16) {
      const int j = jbase + lane;
      const float dx = s_x[j * 3 + 0] - xi0;
      const float dy = s_x[j * 3 + 1] - xi1;
      const float dz = s_x[j * 3 + 2] - xi2;
      const float nrm = fast_sqrt(dx * dx + dy * dy + dz * dz + EPS);
      const float q = qjB[j] + qii;
      const float sem = (q > 0.f) ? q : 0.01f * q;
      s_w[wave * 16 + lane] = sem * __expf(nrm - smax) * invden;
    }

    // phase B: h_e tile + aggregations (each lane owns features f0,f0+1)
#pragma unroll 4
    for (int jj = 0; jj < 16; ++jj) {
      const int j = jbase + jj;
      const float dx = s_x[j * 3 + 0] - xi0;
      const float dy = s_x[j * 3 + 1] - xi1;
      const float dz = s_x[j * 3 + 2] - xi2;
      const float d2 = dx * dx + dy * dy + dz * dz;
      const float n2 = d2 + EPS;
      const float nrm = fast_sqrt(n2);
      const float filt = fast_rcp(nrm + 0.1f);
      const float inv2 = fast_rcp(n2 + EPS);
      const float wjj = s_w[wave * 16 + jj];
      const float* ejr = ejB + (size_t)j * 64 + f0;
      const float z0 = filt * s_wes0[f0] + ejr[0] + s_ei[f0] + s_bes[f0];
      const float z1 = filt * s_wes0[f0 + 1] + ejr[1] + s_ei[f0 + 1] + s_bes[f0 + 1];
      const float he0 = fast_silu(z0);
      const float he1 = fast_silu(z1);
      sheW[jj * 66 + f0] = he0;
      sheW[jj * 66 + f0 + 1] = he1;
      agg0 += wjj * he0;
      agg1 += wjj * he1;
      const float* pjr = pjB + (size_t)j * 64 + f0;
      const float ew0 = fast_tanh(pjr[0] + s_pi[f0] + s_bew[f0]);
      const float ew1 = fast_tanh(pjr[1] + s_pi[f0 + 1] + s_bew[f0 + 1]);
      const float v0 = dx * inv2, v1 = dy * inv2, v2 = dz * inv2;
      att00 += ew0 * v0; att01 += ew0 * v1; att02 += ew0 * v2;
      att10 += ew1 * v0; att11 += ew1 * v1; att12 += ew1 * v2;
    }

    // phase C: A-fragments from LDS, 8x WMMA -> T = h_e @ W_c1
    v16bf A0, A1;
    const float* her = sheW + lo * 66 + 8 * hi;  // row m=lo, khigh folded in
#pragma unroll
    for (int e = 0; e < 16; ++e) {
      const int r = e >> 1;
      const int kk = ((r < 4) ? 0 : 16) + 2 * (r & 3) + (e & 1);
      A0[e] = (__bf16)her[kk];
      A1[e] = (__bf16)her[32 + kk];
    }
    v8f acc[4];
#pragma unroll
    for (int tt = 0; tt < 4; ++tt)
#pragma unroll
      for (int p = 0; p < 8; ++p) acc[tt][p] = 0.f;
#pragma unroll
    for (int tt = 0; tt < 4; ++tt) {
      acc[tt] = __builtin_amdgcn_wmma_f32_16x16x32_bf16(
          false, A0, false, Bf[tt][0], (short)0, acc[tt], false, false);
      acc[tt] = __builtin_amdgcn_wmma_f32_16x16x32_bf16(
          false, A1, false, Bf[tt][1], (short)0, acc[tt], false, false);
    }

    // phase D: coord = silu(T + b_c1) @ W_c2 + b_c2, then x-update partials
    float part[8];
#pragma unroll
    for (int p = 0; p < 8; ++p) part[p] = 0.f;
#pragma unroll
    for (int tt = 0; tt < 4; ++tt) {
      const int f = 16 * tt + lo;
      const float bc = s_bc1[f];
      const float wc = s_wc2[f];
#pragma unroll
      for (int p = 0; p < 8; ++p) {
        part[p] += fast_silu(acc[tt][p] + bc) * wc;
      }
    }
#pragma unroll
    for (int p = 0; p < 8; ++p) {
      part[p] += __shfl_xor(part[p], 1, 32);
      part[p] += __shfl_xor(part[p], 2, 32);
      part[p] += __shfl_xor(part[p], 4, 32);
      part[p] += __shfl_xor(part[p], 8, 32);
    }
    if (lo == 0) {
#pragma unroll
      for (int p = 0; p < 8; ++p)
        s_coord[wave * 16 + 8 * hi + p] = part[p] + bc2v;  // row m = p + 8*hi
    }
    if (lane < 16) {
      const int j = jbase + lane;
      const float co = s_coord[wave * 16 + lane];
      ax0 += co * (s_x[j * 3 + 0] - xi0);
      ax1 += co * (s_x[j * 3 + 1] - xi1);
      ax2 += co * (s_x[j * 3 + 2] - xi2);
    }
  }

  // ---- deterministic cross-wave reduction --------------------------------
  s_aggW[wave * 64 + f0] = agg0;
  s_aggW[wave * 64 + f0 + 1] = agg1;
  s_attW[wave * 192 + f0 * 3 + 0] = att00;
  s_attW[wave * 192 + f0 * 3 + 1] = att01;
  s_attW[wave * 192 + f0 * 3 + 2] = att02;
  s_attW[wave * 192 + (f0 + 1) * 3 + 0] = att10;
  s_attW[wave * 192 + (f0 + 1) * 3 + 1] = att11;
  s_attW[wave * 192 + (f0 + 1) * 3 + 2] = att12;
  if (lane < 16) {
    s_axW[wave * 48 + lane * 3 + 0] = ax0;
    s_axW[wave * 48 + lane * 3 + 1] = ax1;
    s_axW[wave * 48 + lane * 3 + 2] = ax2;
  }
  __syncthreads();

  if (t < 64) {
    float ag = 0.f, a0 = 0.f, a1 = 0.f, a2 = 0.f;
#pragma unroll
    for (int w = 0; w < 4; ++w) {
      ag += s_aggW[w * 64 + t];
      a0 += s_attW[w * 192 + t * 3 + 0];
      a1 += s_attW[w * 192 + t * 3 + 1];
      a2 += s_attW[w * 192 + t * 3 + 2];
    }
    s_agg[t] = ag;
    s_attn[t] = fast_sqrt(fmaxf(a0 * a0 + a1 * a1 + a2 * a2, 0.f) + EPS);
  } else if (t < 67) {
    const int c = t - 64;
    float s = 0.f;
    for (int p = 0; p < 64; ++p) s += s_axW[p * 3 + c];
    const float xic = (c == 0) ? xi0 : ((c == 1) ? xi1 : xi2);
    out[(size_t)BB * NN * 64 + (size_t)node * 3 + c] = s + xic;  // x_new
  }
  __syncthreads();

  // ---- h_new = concat([h_i, h_e_agg, att_norm]) @ W_n + b_n --------------
  if (t < 64) {
    float accn = b_n[t];
#pragma unroll 4
    for (int r = 0; r < 64; ++r) accn += s_hi[r] * W_n[r * 64 + t];
#pragma unroll 4
    for (int r = 0; r < 64; ++r) accn += s_agg[r] * W_n[(64 + r) * 64 + t];
#pragma unroll 4
    for (int r = 0; r < 64; ++r) accn += s_attn[r] * W_n[(128 + r) * 64 + t];
    out[(size_t)node * 64 + t] = accn;  // h_new
  }
}

// ---------------------------------------------------------------------------
extern "C" void kernel_launch(void* const* d_in, const int* in_sizes, int n_in,
                              void* d_out, int out_size, void* d_ws, size_t ws_size,
                              hipStream_t stream) {
  (void)in_sizes; (void)n_in; (void)out_size; (void)ws_size;
  const float* h    = (const float*)d_in[0];
  const float* x    = (const float*)d_in[1];
  const float* W_ew = (const float*)d_in[2];
  const float* b_ew = (const float*)d_in[3];
  const float* W_sa = (const float*)d_in[4];
  const float* W_es = (const float*)d_in[5];
  const float* b_es = (const float*)d_in[6];
  const float* W_c1 = (const float*)d_in[7];
  const float* b_c1 = (const float*)d_in[8];
  const float* W_c2 = (const float*)d_in[9];
  const float* b_c2 = (const float*)d_in[10];
  const float* W_n  = (const float*)d_in[11];
  const float* b_n  = (const float*)d_in[12];
  float* out = (float*)d_out;

  float* ws = (float*)d_ws;  // 528384 floats (~2.02 MB) used
  float* pj = ws;
  float* pi = ws + 131072;
  float* ejv = ws + 262144;
  float* eiv = ws + 393216;
  float* qjv = ws + 524288;
  float* qiv = ws + 526336;

  sake_node_proj<<<dim3(BB * NN), dim3(64), 0, stream>>>(
      h, W_ew, W_sa, W_es, pj, pi, ejv, eiv, qjv, qiv);
  sake_main<<<dim3(BB * NN), dim3(128), 0, stream>>>(
      h, x, b_ew, W_es, b_es, W_c1, b_c1, W_c2, b_c2, W_n, b_n,
      pj, pi, ejv, eiv, qjv, qiv, out);
}
